// GRU_GNN_Model_10874857193921
// MI455X (gfx1250) — matrix operand order repaired and checked by reading, not compiled, not run on hardware
//
#include <hip/hip_runtime.h>
#include <hip/hip_bf16.h>
#include <math.h>

#define BB 16
#define VV 4096
#define EE 24576
#define INF_ 256
#define HH 512
#define G1C 256
#define G2C 512
#define GOD 16
#define OUT2 (VV*GOD)

typedef __attribute__((ext_vector_type(16))) __bf16 v16bf;
typedef __attribute__((ext_vector_type(8)))  __bf16 v8bf;
typedef __attribute__((ext_vector_type(8)))  float  v8f;

__device__ __forceinline__ unsigned short f2bf(float f) {
    unsigned u = __float_as_uint(f);
    unsigned r = (u + 0x7FFFu + ((u >> 16) & 1u)) >> 16;
    return (unsigned short)r;
}

__device__ __forceinline__ void atomicMaxF32(float* p, float v) {
    asm volatile("global_atomic_max_num_f32 %0, %1, off scope:SCOPE_DEV"
                 :: "v"(p), "v"(v) : "memory");
}

// A fragment (16x32 bf16) from a row-major 16x512 ushort tile in LDS, contiguous k order (k = 16h + s)
__device__ __forceinline__ v16bf load_a_frag(const unsigned short* lds, int lane, int k0) {
    int row = lane & 15;
    int kb  = (lane >> 4) << 4;            // 0 or 16
    const unsigned short* p0 = lds + row * 512 + k0 + kb;
    v8bf lo = *(const v8bf*)p0;
    v8bf hi = *(const v8bf*)(p0 + 8);
    v16bf a;
#pragma unroll
    for (int i = 0; i < 8; ++i) { a[i] = lo[i]; a[8 + i] = hi[i]; }
    return a;
}

// B fragment (32x16 bf16) from row-major weight W[g][k], ldk = 512
__device__ __forceinline__ v16bf load_b_frag(const unsigned short* w, int lane, int gbase, int k0) {
    int col = lane & 15;
    int kb  = (lane >> 4) << 4;            // 0 or 16
    return *(const v16bf*)(w + (size_t)(gbase + col) * 512 + k0 + kb);
}

// ---------------- utility kernels ----------------
__global__ void init_neg_kernel(float* p, long long n) {
    long long i = (long long)blockIdx.x * blockDim.x + threadIdx.x;
    long long stride = (long long)gridDim.x * blockDim.x;
    for (; i < n; i += stride) p[i] = -3.0e38f;
}

__global__ void cvt_bf16_kernel(const float* __restrict__ in, unsigned short* __restrict__ out, long long n) {
    long long i = (long long)blockIdx.x * blockDim.x + threadIdx.x;
    long long stride = (long long)gridDim.x * blockDim.x;
    for (; i < n; i += stride) out[i] = f2bf(in[i]);
}

// ---------------- edge conv 1 (tiny) ----------------
__global__ __launch_bounds__(256)
void conv1_kernel(const float* __restrict__ svp, const int* __restrict__ edge,
                  const float* __restrict__ We1, const float* __restrict__ be1,
                  const float* __restrict__ ae1, float* __restrict__ x0) {
    int e = blockIdx.x, b = blockIdx.y, g = threadIdx.x;
    int vd = edge[EE + e], vs = edge[e];
    const float* pi = svp + ((size_t)b * VV + vd) * 3;
    const float* pj = svp + ((size_t)b * VV + vs) * 3;
    float m0 = pi[0], m1 = pi[1], m2 = pi[2];
    float m3 = pj[0] - m0, m4 = pj[1] - m1, m5 = pj[2] - m2;
    const float* w = We1 + g * 6;
    float v = m0*w[0] + m1*w[1] + m2*w[2] + m3*w[3] + m4*w[4] + m5*w[5] + be1[g];
    v = (v >= 0.f) ? v : ae1[g] * v;
    atomicMaxF32(&x0[((size_t)b * VV + vd) * G1C + g], v);
}

// ---------------- edge conv 2: dominant WMMA matmul ----------------
// 64 edges per WG (4 sub-tiles of 16) so each B fragment feeds 4 WMMAs.
__global__ __launch_bounds__(256)
void conv2_kernel(const float* __restrict__ x0, const unsigned short* __restrict__ We2bf,
                  const float* __restrict__ be2, const float* __restrict__ ae2,
                  const int* __restrict__ edge, float* __restrict__ x1) {
    __shared__ unsigned short sm[64 * 512];   // 64 KB message tile (bf16)
    __shared__ int sdst[64];
    __shared__ int ssrc[64];
    const int et = blockIdx.x, b = blockIdx.y, t = threadIdx.x;
    const int e0 = et * 64;

    if (t < 64) sdst[t] = edge[EE + e0 + t];
    else if (t < 128) ssrc[t - 64] = edge[e0 + (t - 64)];
    __syncthreads();

    // build message m = [xi, xj - xi] for 64 edges, bf16, in LDS
    for (int idx = t; idx < 64 * 512; idx += 256) {
        int e = idx >> 9, f = idx & 511;
        int vd = sdst[e], vs = ssrc[e];
        float val;
        if (f < 256) {
            val = x0[((size_t)b * VV + vd) * G1C + f];
        } else {
            int ff = f - 256;
            val = x0[((size_t)b * VV + vs) * G1C + ff] - x0[((size_t)b * VV + vd) * G1C + ff];
        }
        sm[idx] = f2bf(val);
    }
    __syncthreads();

    const int wave = t >> 5, lane = t & 31;
    for (int nt = wave; nt < 32; nt += 8) {       // wave-uniform loop, EXEC all ones
        int gbase = nt * 16;
        v8f c0 = {0.f,0.f,0.f,0.f,0.f,0.f,0.f,0.f};
        v8f c1 = c0, c2 = c0, c3 = c0;
#pragma unroll
        for (int k0 = 0; k0 < 512; k0 += 32) {
            v16bf bm = load_b_frag(We2bf, lane, gbase, k0);
            v16bf a0 = load_a_frag(sm + 0 * 16 * 512, lane, k0);
            v16bf a1 = load_a_frag(sm + 1 * 16 * 512, lane, k0);
            v16bf a2 = load_a_frag(sm + 2 * 16 * 512, lane, k0);
            v16bf a3 = load_a_frag(sm + 3 * 16 * 512, lane, k0);
            c0 = __builtin_amdgcn_wmma_f32_16x16x32_bf16(false, a0, false, bm, (short)0, c0, false, false);
            c1 = __builtin_amdgcn_wmma_f32_16x16x32_bf16(false, a1, false, bm, (short)0, c1, false, false);
            c2 = __builtin_amdgcn_wmma_f32_16x16x32_bf16(false, a2, false, bm, (short)0, c2, false, false);
            c3 = __builtin_amdgcn_wmma_f32_16x16x32_bf16(false, a3, false, bm, (short)0, c3, false, false);
        }
        int n = gbase + (lane & 15);
        float bee = be2[n], aee = ae2[n];
        int mof = (lane >> 4) << 3;
#pragma unroll
        for (int i = 0; i < 8; ++i) {
            int er = i + mof;
            float v0 = c0[i] + bee; v0 = (v0 >= 0.f) ? v0 : aee * v0;
            float v1 = c1[i] + bee; v1 = (v1 >= 0.f) ? v1 : aee * v1;
            float v2 = c2[i] + bee; v2 = (v2 >= 0.f) ? v2 : aee * v2;
            float v3 = c3[i] + bee; v3 = (v3 >= 0.f) ? v3 : aee * v3;
            atomicMaxF32(&x1[((size_t)b * VV + sdst[er])      * G2C + n], v0);
            atomicMaxF32(&x1[((size_t)b * VV + sdst[16 + er]) * G2C + n], v1);
            atomicMaxF32(&x1[((size_t)b * VV + sdst[32 + er]) * G2C + n], v2);
            atomicMaxF32(&x1[((size_t)b * VV + sdst[48 + er]) * G2C + n], v3);
        }
    }
}

// ---------------- GRU cell (small) ----------------
__global__ __launch_bounds__(256)
void gru_kernel(const float* __restrict__ x, const float* __restrict__ h,
                const float* __restrict__ Wih, const float* __restrict__ Whh,
                const float* __restrict__ bih, const float* __restrict__ bhh,
                float* __restrict__ nh, float* __restrict__ out_nh) {
    int wid = (blockIdx.x * 256 + threadIdx.x) >> 5;  // 0..8191
    int lane = threadIdx.x & 31;
    int b = wid >> 9, j = wid & 511;
    const float* xb = x + (size_t)b * INF_;
    const float* hb = h + (size_t)b * HH;
    float s_r = 0.f, s_z = 0.f, s_n = 0.f, s_hn = 0.f;
    for (int f = lane; f < INF_; f += 32) {
        float xv = xb[f];
        s_r += xv * Wih[(size_t)j * INF_ + f];
        s_z += xv * Wih[(size_t)(HH + j) * INF_ + f];
        s_n += xv * Wih[(size_t)(2 * HH + j) * INF_ + f];
    }
    for (int f = lane; f < HH; f += 32) {
        float hv = hb[f];
        s_r  += hv * Whh[(size_t)j * HH + f];
        s_z  += hv * Whh[(size_t)(HH + j) * HH + f];
        s_hn += hv * Whh[(size_t)(2 * HH + j) * HH + f];
    }
    for (int m = 16; m >= 1; m >>= 1) {
        s_r += __shfl_xor(s_r, m); s_z += __shfl_xor(s_z, m);
        s_n += __shfl_xor(s_n, m); s_hn += __shfl_xor(s_hn, m);
    }
    if (lane == 0) {
        float ir  = s_r + bih[j] + bhh[j];
        float iz  = s_z + bih[HH + j] + bhh[HH + j];
        float inn = s_n + bih[2 * HH + j];
        float hn  = s_hn + bhh[2 * HH + j];
        float r = 1.f / (1.f + expf(-ir));
        float z = 1.f / (1.f + expf(-iz));
        float n = tanhf(inn + r * hn);
        float hv = hb[j];
        float o = (1.f - z) * n + z * hv;
        nh[(size_t)b * HH + j] = o;
        out_nh[(size_t)b * HH + j] = o;
    }
}

// ---------------- W1 + PReLU + BN1 -> g1 (bf16) ----------------
__global__ __launch_bounds__(256)
void w1bn_kernel(const float* __restrict__ nh, const float* __restrict__ W1,
                 const float* __restrict__ b1, const float* __restrict__ a1,
                 const float* __restrict__ g, const float* __restrict__ bt,
                 const float* __restrict__ mm, const float* __restrict__ vv,
                 unsigned short* __restrict__ g1bf) {
    int wid = (blockIdx.x * 256 + threadIdx.x) >> 5;
    int lane = threadIdx.x & 31;
    int b = wid >> 9, j = wid & 511;
    float s = 0.f;
    const float* nb = nh + (size_t)b * HH;
    for (int f = lane; f < HH; f += 32) s += nb[f] * W1[(size_t)j * HH + f];
    for (int m = 16; m >= 1; m >>= 1) s += __shfl_xor(s, m);
    if (lane == 0) {
        float tt = s + b1[j];
        tt = (tt >= 0.f) ? tt : a1[j] * tt;
        tt = (tt - mm[j]) * rsqrtf(vv[j] + 1e-5f) * g[j] + bt[j];
        g1bf[(size_t)b * HH + j] = f2bf(tt);
    }
}

// ---------------- g1 @ W2^T + PReLU + BN2 (bandwidth-bound, bf16 WMMA) ----------------
__global__ __launch_bounds__(256)
void w2_kernel(const unsigned short* __restrict__ g1bf, const unsigned short* __restrict__ W2bf,
               const float* __restrict__ b2, const float* __restrict__ a2,
               const float* __restrict__ g, const float* __restrict__ bt,
               const float* __restrict__ mm, const float* __restrict__ vv,
               float* __restrict__ gru_out) {
    __shared__ unsigned short sA[16 * 512];
    int t = threadIdx.x;
    for (int idx = t; idx < 16 * 512; idx += 256) sA[idx] = g1bf[idx];
    __syncthreads();
    int wave = t >> 5, lane = t & 31;
    int gbase = (blockIdx.x * 8 + wave) * 16;
    v8f c = {0.f,0.f,0.f,0.f,0.f,0.f,0.f,0.f};
#pragma unroll
    for (int k0 = 0; k0 < 512; k0 += 32) {
        v16bf a  = load_a_frag(sA, lane, k0);
        v16bf bm = load_b_frag(W2bf, lane, gbase, k0);
        c = __builtin_amdgcn_wmma_f32_16x16x32_bf16(false, a, false, bm,
                                                    (short)0, c, false, false);
    }
    int n = gbase + (lane & 15);
    float b2v = b2[n], a2v = a2[n];
    float sc = rsqrtf(vv[n] + 1e-5f) * g[n], mb = mm[n], btv = bt[n];
    int mof = (lane >> 4) << 3;
#pragma unroll
    for (int i = 0; i < 8; ++i) {
        int batch = i + mof;
        float xv = c[i] + b2v;
        xv = (xv >= 0.f) ? xv : a2v * xv;
        xv = (xv - mb) * sc + btv;
        gru_out[(size_t)batch * OUT2 + n] = xv;
    }
}

// ---------------- final projection: concat(gru_out, x1) @ Wo^T + bo ----------------
__global__ __launch_bounds__(256)
void out_kernel(const float* __restrict__ gru_out, const float* __restrict__ x1,
                const float* __restrict__ Wo, const float* __restrict__ bo,
                float* __restrict__ y) {
    int wid = (blockIdx.x * 256 + threadIdx.x) >> 5;  // 0..65535
    int lane = threadIdx.x & 31;
    int b = wid >> 12, v = wid & 4095;
    const float* xr = x1 + ((size_t)b * VV + v) * G2C;
    const float* gr = gru_out + (size_t)b * OUT2 + (size_t)v * GOD;
    float a0 = 0.f, a1 = 0.f, a2 = 0.f;
    for (int idx = lane; idx < 528; idx += 32) {
        float f = (idx < 16) ? gr[idx] : xr[idx - 16];
        a0 += f * Wo[idx];
        a1 += f * Wo[528 + idx];
        a2 += f * Wo[1056 + idx];
    }
    for (int m = 16; m >= 1; m >>= 1) {
        a0 += __shfl_xor(a0, m); a1 += __shfl_xor(a1, m); a2 += __shfl_xor(a2, m);
    }
    if (lane == 0) {
        size_t o = ((size_t)b * VV + v) * 3;
        y[o] = a0 + bo[0]; y[o + 1] = a1 + bo[1]; y[o + 2] = a2 + bo[2];
    }
}

extern "C" void kernel_launch(void* const* d_in, const int* in_sizes, int n_in,
                              void* d_out, int out_size, void* d_ws, size_t ws_size,
                              hipStream_t stream) {
    const float* x    = (const float*)d_in[0];
    const float* svp  = (const float*)d_in[1];
    const float* hid  = (const float*)d_in[2];
    const int*   edge = (const int*)  d_in[3];
    const float* Wih  = (const float*)d_in[4];
    const float* Whh  = (const float*)d_in[5];
    const float* bih  = (const float*)d_in[6];
    const float* bhh  = (const float*)d_in[7];
    const float* W1   = (const float*)d_in[8];
    const float* b1   = (const float*)d_in[9];
    const float* a1   = (const float*)d_in[10];
    const float* bn1g = (const float*)d_in[11];
    const float* bn1b = (const float*)d_in[12];
    const float* bn1m = (const float*)d_in[13];
    const float* bn1v = (const float*)d_in[14];
    const float* W2   = (const float*)d_in[15];
    const float* b2   = (const float*)d_in[16];
    const float* a2   = (const float*)d_in[17];
    const float* bn2g = (const float*)d_in[18];
    const float* bn2b = (const float*)d_in[19];
    const float* bn2m = (const float*)d_in[20];
    const float* bn2v = (const float*)d_in[21];
    const float* Wo   = (const float*)d_in[22];
    const float* bo   = (const float*)d_in[23];
    const float* We1  = (const float*)d_in[24];
    const float* be1  = (const float*)d_in[25];
    const float* ae1  = (const float*)d_in[26];
    const float* We2  = (const float*)d_in[27];
    const float* be2  = (const float*)d_in[28];
    const float* ae2  = (const float*)d_in[29];

    char* w = (char*)d_ws;
    float*          x0    = (float*)(w);                         //  67,108,864 B
    float*          x1    = (float*)(w + 67108864ull);           // 134,217,728 B
    unsigned short* We2bf = (unsigned short*)(w + 201326592ull); //     524,288 B
    unsigned short* W2bf  = (unsigned short*)(w + 201850880ull); //  67,108,864 B
    float*          nh    = (float*)(w + 268959744ull);          //      32,768 B
    unsigned short* g1bf  = (unsigned short*)(w + 268992512ull); //      16,384 B
    float*          gout  = (float*)(w + 269008896ull);          //   4,194,304 B

    float* y      = (float*)d_out;                 // B*V*3 = 196608
    float* out_nh = (float*)d_out + 196608;        // B*H   =   8192

    // 1) init segment-max accumulators (x0 then x1, contiguous region)
    init_neg_kernel<<<8192, 256, 0, stream>>>(x0, (long long)BB * VV * (G1C + G2C));
    // 2) weight conversions to bf16
    cvt_bf16_kernel<<<1024, 256, 0, stream>>>(We2, We2bf, (long long)G2C * 512);
    cvt_bf16_kernel<<<8192, 256, 0, stream>>>(W2, W2bf, (long long)OUT2 * HH);
    // 3) edge conv 1
    conv1_kernel<<<dim3(EE, BB), 256, 0, stream>>>(svp, edge, We1, be1, ae1, x0);
    // 4) edge conv 2 (dominant WMMA matmul + segment max), 64 edges / WG
    conv2_kernel<<<dim3(EE / 64, BB), 256, 0, stream>>>(x0, We2bf, be2, ae2, edge, x1);
    // 5) GRU cell
    gru_kernel<<<1024, 256, 0, stream>>>(x, hid, Wih, Whh, bih, bhh, nh, out_nh);
    // 6) W1 + PReLU + BN1 -> bf16
    w1bn_kernel<<<1024, 256, 0, stream>>>(nh, W1, b1, a1, bn1g, bn1b, bn1m, bn1v, g1bf);
    // 7) W2 (bandwidth-bound) via bf16 WMMA + PReLU + BN2
    w2_kernel<<<512, 256, 0, stream>>>(g1bf, W2bf, b2, a2, bn2g, bn2b, bn2m, bn2v, gout);
    // 8) final projection
    out_kernel<<<8192, 256, 0, stream>>>(gout, x1, Wo, bo, y);
}
